// PointerNet_65747359367241
// MI455X (gfx1250) — matrix operand order repaired and checked by reading, not compiled, not run on hardware
//
#include <hip/hip_runtime.h>
#include <hip/hip_bf16.h>
#include <stdint.h>

#define BB 32
#define TT 2048
#define HH 256
#define GG 1024   /* 4*H */
#define MM 64
#define LL 256

typedef __attribute__((ext_vector_type(16))) __bf16 v16bf;
typedef __attribute__((ext_vector_type(8)))  float  v8f;

union Frag { uint32_t u[8]; v16bf v; };

__device__ __forceinline__ uint16_t f2bf(float f) {
    uint32_t u = __float_as_uint(f);
    u += 0x7FFFu + ((u >> 16) & 1u);   // round-to-nearest-even
    return (uint16_t)(u >> 16);
}

__device__ __forceinline__ v8f vzero8() {
    v8f z = {0.f, 0.f, 0.f, 0.f, 0.f, 0.f, 0.f, 0.f};
    return z;
}

__device__ __forceinline__ v8f wmma_bf16(v16bf a, v16bf b, v8f c) {
    return __builtin_amdgcn_wmma_f32_16x16x32_bf16(false, a, false, b, (short)0, c, false, false);
}

// One fragment = 8 consecutive dwords per lane (pre-packed layout), two b128 loads.
__device__ __forceinline__ v16bf load_frag_global(const uint32_t* p) {
    Frag f;
    const uint4* q = (const uint4*)p;
    uint4 lo = q[0];
    uint4 hi = q[1];
    f.u[0] = lo.x; f.u[1] = lo.y; f.u[2] = lo.z; f.u[3] = lo.w;
    f.u[4] = hi.x; f.u[5] = hi.y; f.u[6] = hi.z; f.u[7] = hi.w;
    return f.v;
}

// Build 16x32 bf16 A fragment from LDS hidden state ([32 rows][128 dwords], k-pairs packed).
// Layout: lane<16: M=lane, K = {0..15} in dw0-7; lane>=16: K += 16.
__device__ __forceinline__ v16bf load_afrag_lds(const uint32_t* h_u32, int row, int kstep, int lane) {
    int kbase = kstep * 32 + ((lane & 16) ? 16 : 0);
    Frag f;
    const uint32_t* rp = h_u32 + row * 128;
#pragma unroll
    for (int d = 0; d < 8; ++d) {
        int k = kbase + 2 * d;
        f.u[d] = rp[k >> 1];
    }
    return f.v;
}

__device__ __forceinline__ float fsigmoid(float x) { return 1.0f / (1.0f + __expf(-x)); }
__device__ __forceinline__ float ftanh_(float x) {
    float e = __expf(2.0f * x);
    return (e - 1.0f) / (e + 1.0f);
}

// ---------------------------------------------------------------------------
// Prep: fold input projection (rank-2) into per-gate coefficients; fuse biases.
// ---------------------------------------------------------------------------
__global__ void prep_misc(const float* __restrict__ Wp, const float* __restrict__ bp,
                          const float* __restrict__ W_ih_e, const float* __restrict__ b_ih_e,
                          const float* __restrict__ b_hh_e,
                          const float* __restrict__ b_ih_d, const float* __restrict__ b_hh_d,
                          float* __restrict__ Weff, float* __restrict__ beff, float* __restrict__ bd) {
    int j = blockIdx.x * blockDim.x + threadIdx.x;
    if (j >= GG) return;
    float w0 = 0.f, w1 = 0.f, b = 0.f;
    for (int e = 0; e < HH; ++e) {      // E == 256
        float wie = W_ih_e[j * HH + e];
        w0 += Wp[e] * wie;              // Wp[0][e]
        w1 += Wp[HH + e] * wie;         // Wp[1][e]
        b  += bp[e] * wie;
    }
    Weff[j]      = w0;
    Weff[GG + j] = w1;
    beff[j] = b + b_ih_e[j] + b_hh_e[j];
    bd[j]   = b_ih_d[j] + b_hh_d[j];
}

// ---------------------------------------------------------------------------
// Pack weights into bf16 WMMA B-fragment layout:
//   dword index = ((nt*ksteps + ks)*32 + lane)*8 + d
//   lane: n = nt*16 + (lane&15); k = ks*32 + (lane>=16 ? 16 : 0) + 2*d (+0,+1)
// mode 0: W(k,n) = A[n*K + k] (+A2)   (torch [4H,H] layout, B = W^T)
// mode 1: W(k,n) = A[k*N + n]         (Wk/Wq already [K,N])
// ---------------------------------------------------------------------------
__global__ void pack_b(const float* __restrict__ A, const float* __restrict__ A2,
                       uint32_t* __restrict__ dst, int K, int N, int mode) {
    int idx = blockIdx.x * blockDim.x + threadIdx.x;
    int ksteps = K >> 5;
    int total = (N >> 4) * ksteps * 256;
    if (idx >= total) return;
    int d    = idx & 7;
    int lane = (idx >> 3) & 31;
    int ks   = (idx >> 8) % ksteps;
    int nt   = (idx >> 8) / ksteps;
    int n = nt * 16 + (lane & 15);
    int k = ks * 32 + ((lane & 16) ? 16 : 0) + 2 * d;
    float lo, hi;
    if (mode == 0) {
        lo = A[n * K + k];
        hi = A[n * K + k + 1];
        if (A2) { lo += A2[n * K + k]; hi += A2[n * K + k + 1]; }
    } else {
        lo = A[k * N + n];
        hi = A[(k + 1) * N + n];
    }
    dst[idx] = (uint32_t)f2bf(lo) | ((uint32_t)f2bf(hi) << 16);
}

// ---------------------------------------------------------------------------
// Encoder: persistent single workgroup, 32 waves. Per step:
//   all 32 waves: gates = h @ W_hh^T via v_wmma_f32_16x16x32_bf16 (wave owns
//   1 M-tile x 4 gate N-tiles of one 16-col hidden slice -> i/f/g/o in-lane).
//   waves 24..31: fused kproj = enc(t-1) @ Wk + bk, scattered straight into
//   the scores-GEMM B-fragment layout. hT/cT captured at t == seq_len-1.
// ---------------------------------------------------------------------------
__global__ __launch_bounds__(1024) void encoder_kernel(
    const float* __restrict__ seq, const int* __restrict__ seq_lens,
    const uint32_t* __restrict__ WBe, const float* __restrict__ Weff, const float* __restrict__ beff,
    const uint32_t* __restrict__ WBk, const float* __restrict__ bk,
    uint16_t* __restrict__ kpP, float* __restrict__ hT, float* __restrict__ cT) {
    __shared__ uint32_t h_lds[2][BB * 128];   // bf16 pairs, double buffered (32 KB)
    __shared__ float seq_lds[2][BB * 2];
    __shared__ int sl_lds[BB];

    const int tid  = threadIdx.x;
    const int wave = tid >> 5;
    const int lane = tid & 31;
    const int mt    = wave >> 4;                       // 0..1 (batch tile)
    const int jht   = wave & 15;                       // hidden-col tile
    const int jh    = jht * 16 + (lane & 15);
    const int rowA  = mt * 16 + (lane & 15);
    const int bbase = mt * 16 + ((lane >> 4) << 3);    // first of 8 batch rows this lane owns

    for (int i = tid; i < BB * 128; i += 1024) h_lds[0][i] = 0u;
    if (tid < BB) sl_lds[tid] = seq_lens[tid];
    if (tid < 2 * BB) {
        int b = tid >> 1, c = tid & 1;
        seq_lds[0][tid] = seq[(size_t)b * TT * 2 + c];
    }
    __syncthreads();

    float we0[4], we1[4], be[4];
#pragma unroll
    for (int g = 0; g < 4; ++g) {
        int n = g * 256 + jh;
        we0[g] = Weff[n];
        we1[g] = Weff[GG + n];
        be[g]  = beff[n];
    }
    int slv[8];
#pragma unroll
    for (int v = 0; v < 8; ++v) slv[v] = sl_lds[bbase + v];

    // kproj role (waves 24..31): 2 M-tiles x 4 N-tiles of [32 x 64]
    const int kw      = wave - 24;
    const int kp_mt   = (kw >> 2) & 1;
    const int kp_nt   = kw & 3;
    const int kp_rowA = kp_mt * 16 + (lane & 15);
    const int kp_b0   = kp_mt * 16 + ((lane >> 4) << 3);
    const int m_col   = kp_nt * 16 + (lane & 15);
    const int kp_ks   = m_col >> 5;
    const int km      = m_col & 31;
    const int kp_lhi  = (km >= 16) ? 16 : 0;
    const int kadj    = km - kp_lhi;
    const int kp_d    = kadj >> 1;
    const int kp_half = kadj & 1;
    float bkm = 0.f;
    int slk[8];
    if (wave >= 24) {
        bkm = bk[m_col];
#pragma unroll
        for (int v = 0; v < 8; ++v) slk[v] = sl_lds[kp_b0 + v];
    }

    float c_state[8];
#pragma unroll
    for (int v = 0; v < 8; ++v) c_state[v] = 0.f;

    for (int t = 0; t <= TT; ++t) {
        const int cur = t & 1, nxt = cur ^ 1;

        if (t < TT) {
            if (tid < 2 * BB && (t + 1) < TT) {        // stage inputs for next step
                int b = tid >> 1, c = tid & 1;
                seq_lds[nxt][tid] = seq[(size_t)b * TT * 2 + (size_t)(t + 1) * 2 + c];
            }

            v8f acc0 = vzero8(), acc1 = vzero8(), acc2 = vzero8(), acc3 = vzero8();
#pragma unroll
            for (int ks = 0; ks < 8; ++ks) {
                v16bf a = load_afrag_lds(&h_lds[cur][0], rowA, ks, lane);
                acc0 = wmma_bf16(a, load_frag_global(WBe + ((((0 * 16 + jht) * 8 + ks) * 32 + lane) * 8)), acc0);
                acc1 = wmma_bf16(a, load_frag_global(WBe + ((((1 * 16 + jht) * 8 + ks) * 32 + lane) * 8)), acc1);
                acc2 = wmma_bf16(a, load_frag_global(WBe + ((((2 * 16 + jht) * 8 + ks) * 32 + lane) * 8)), acc2);
                acc3 = wmma_bf16(a, load_frag_global(WBe + ((((3 * 16 + jht) * 8 + ks) * 32 + lane) * 8)), acc3);
            }

            uint16_t* hn = (uint16_t*)&h_lds[nxt][0];
#pragma unroll
            for (int v = 0; v < 8; ++v) {
                int b = bbase + v;
                float x0 = seq_lds[cur][b * 2 + 0];
                float x1 = seq_lds[cur][b * 2 + 1];
                float gi = acc0[v] + x0 * we0[0] + x1 * we1[0] + be[0];
                float gf = acc1[v] + x0 * we0[1] + x1 * we1[1] + be[1];
                float gg = acc2[v] + x0 * we0[2] + x1 * we1[2] + be[2];
                float go = acc3[v] + x0 * we0[3] + x1 * we1[3] + be[3];
                float cn = fsigmoid(gf) * c_state[v] + fsigmoid(gi) * ftanh_(gg);
                float hf = fsigmoid(go) * ftanh_(cn);
                c_state[v] = cn;
                hn[b * HH + jh] = f2bf(hf);
                if (t == slv[v] - 1) {
                    hT[b * HH + jh] = hf;
                    cT[b * HH + jh] = cn;
                }
            }
        }

        if (t >= 1 && wave >= 24) {                    // kproj for enc step t-1
            const int te = t - 1;
            const int tt = te >> 4, tc = te & 15;
            v8f kacc = vzero8();
#pragma unroll
            for (int ks = 0; ks < 8; ++ks) {
                v16bf a = load_afrag_lds(&h_lds[cur][0], kp_rowA, ks, lane);
                kacc = wmma_bf16(a, load_frag_global(WBk + (((kp_nt * 8 + ks) * 32 + lane) * 8)), kacc);
            }
            const int laneB = tc + kp_lhi;
#pragma unroll
            for (int v = 0; v < 8; ++v) {
                int b = kp_b0 + v;
                float val = (te < slk[v]) ? (kacc[v] + bkm) : bkm;  // masked enc -> bk only
                size_t dwI = ((((size_t)b * 128 + tt) * 2 + kp_ks) * 32 + laneB) * 8 + kp_d;
                kpP[dwI * 2 + kp_half] = f2bf(val);
            }
        }

        __syncthreads();
    }
}

// ---------------------------------------------------------------------------
// Decoder: same machinery; x == h so weights pre-summed (WBd). Waves 24..31
// compute q = h2 @ Wq + bq for the previous step and store bf16 Q[b][l][m].
// ---------------------------------------------------------------------------
__global__ __launch_bounds__(1024) void decoder_kernel(
    const uint32_t* __restrict__ WBd, const float* __restrict__ bd,
    const uint32_t* __restrict__ WBq, const float* __restrict__ bq,
    const float* __restrict__ hT, const float* __restrict__ cT,
    uint16_t* __restrict__ Qout) {
    __shared__ uint32_t h_lds[2][BB * 128];

    const int tid  = threadIdx.x;
    const int wave = tid >> 5;
    const int lane = tid & 31;
    const int mt    = wave >> 4;
    const int jht   = wave & 15;
    const int jh    = jht * 16 + (lane & 15);
    const int rowA  = mt * 16 + (lane & 15);
    const int bbase = mt * 16 + ((lane >> 4) << 3);

    for (int i = tid; i < BB * 128; i += 1024) {
        float lo = hT[2 * i], hi = hT[2 * i + 1];
        h_lds[0][i] = (uint32_t)f2bf(lo) | ((uint32_t)f2bf(hi) << 16);
    }

    float bdv[4];
#pragma unroll
    for (int g = 0; g < 4; ++g) bdv[g] = bd[g * 256 + jh];

    float c_state[8];
#pragma unroll
    for (int v = 0; v < 8; ++v) c_state[v] = cT[(bbase + v) * HH + jh];

    const int kw      = wave - 24;
    const int q_mt    = (kw >> 2) & 1;
    const int q_nt    = kw & 3;
    const int q_rowA  = q_mt * 16 + (lane & 15);
    const int q_b0    = q_mt * 16 + ((lane >> 4) << 3);
    const int q_m     = q_nt * 16 + (lane & 15);
    const float bqm   = (wave >= 24) ? bq[q_m] : 0.f;

    __syncthreads();

    for (int l = 0; l <= LL; ++l) {
        const int cur = l & 1, nxt = cur ^ 1;

        if (l < LL) {
            v8f acc0 = vzero8(), acc1 = vzero8(), acc2 = vzero8(), acc3 = vzero8();
#pragma unroll
            for (int ks = 0; ks < 8; ++ks) {
                v16bf a = load_afrag_lds(&h_lds[cur][0], rowA, ks, lane);
                acc0 = wmma_bf16(a, load_frag_global(WBd + ((((0 * 16 + jht) * 8 + ks) * 32 + lane) * 8)), acc0);
                acc1 = wmma_bf16(a, load_frag_global(WBd + ((((1 * 16 + jht) * 8 + ks) * 32 + lane) * 8)), acc1);
                acc2 = wmma_bf16(a, load_frag_global(WBd + ((((2 * 16 + jht) * 8 + ks) * 32 + lane) * 8)), acc2);
                acc3 = wmma_bf16(a, load_frag_global(WBd + ((((3 * 16 + jht) * 8 + ks) * 32 + lane) * 8)), acc3);
            }
            uint16_t* hn = (uint16_t*)&h_lds[nxt][0];
#pragma unroll
            for (int v = 0; v < 8; ++v) {
                int b = bbase + v;
                float gi = acc0[v] + bdv[0];
                float gf = acc1[v] + bdv[1];
                float gg = acc2[v] + bdv[2];
                float go = acc3[v] + bdv[3];
                float cn = fsigmoid(gf) * c_state[v] + fsigmoid(gi) * ftanh_(gg);
                float hf = fsigmoid(go) * ftanh_(cn);
                c_state[v] = cn;
                hn[b * HH + jh] = f2bf(hf);
            }
        }

        if (l >= 1 && wave >= 24) {                    // q for decoder step l-1
            v8f qacc = vzero8();
#pragma unroll
            for (int ks = 0; ks < 8; ++ks) {
                v16bf a = load_afrag_lds(&h_lds[cur][0], q_rowA, ks, lane);
                qacc = wmma_bf16(a, load_frag_global(WBq + (((q_nt * 8 + ks) * 32 + lane) * 8)), qacc);
            }
#pragma unroll
            for (int v = 0; v < 8; ++v) {
                int b = q_b0 + v;
                Qout[((size_t)b * LL + (l - 1)) * MM + q_m] = f2bf(qacc[v] + bqm);
            }
        }

        __syncthreads();
    }
}

// ---------------------------------------------------------------------------
// Scores: out[b,l,t] = scale * (Q_b @ kproj_b^T) + mask bias.  Pure WMMA:
// A = Q_b rows (bf16, contiguous k pairs), B = pre-packed kproj fragments.
// grid (8 t-blocks, 32 batches), 32 waves: wave = (thalf, l-tile), 8 t-tiles.
// ---------------------------------------------------------------------------
__global__ __launch_bounds__(1024) void scores_kernel(
    const uint16_t* __restrict__ Q, const uint16_t* __restrict__ kpPu,
    const int* __restrict__ seq_lens, float* __restrict__ out) {
    const uint32_t* Qu  = (const uint32_t*)Q;
    const uint32_t* kpP = (const uint32_t*)kpPu;
    const int b     = blockIdx.y;
    const int tb    = blockIdx.x;
    const int tid   = threadIdx.x;
    const int wave  = tid >> 5, lane = tid & 31;
    const int lt    = wave & 15;
    const int thalf = wave >> 4;
    const int slen  = seq_lens[b];
    const int rowQ  = lt * 16 + (lane & 15);

    Frag fa[2];                                        // K = 64 -> 2 ksteps
#pragma unroll
    for (int ks = 0; ks < 2; ++ks) {
        int kbase = ks * 32 + ((lane & 16) ? 16 : 0);
#pragma unroll
        for (int d = 0; d < 8; ++d) {
            int k = kbase + 2 * d;
            fa[ks].u[d] = Qu[((size_t)b * LL + rowQ) * 32 + (k >> 1)];
        }
    }

#pragma unroll
    for (int ttl = 0; ttl < 8; ++ttl) {
        int tt = tb * 16 + thalf * 8 + ttl;
        v8f acc = vzero8();
        acc = wmma_bf16(fa[0].v, load_frag_global(kpP + ((((size_t)b * 128 + tt) * 2 + 0) * 32 + lane) * 8), acc);
        acc = wmma_bf16(fa[1].v, load_frag_global(kpP + ((((size_t)b * 128 + tt) * 2 + 1) * 32 + lane) * 8), acc);
        int tcol = tt * 16 + (lane & 15);
        float bias = (tcol < slen) ? 0.f : -1e9f;
#pragma unroll
        for (int v = 0; v < 8; ++v) {
            int lrow = lt * 16 + ((lane >> 4) << 3) + v;
            out[((size_t)b * LL + lrow) * TT + tcol] = acc[v] * 0.125f + bias;   // 1/sqrt(64)
        }
    }
}

// ---------------------------------------------------------------------------
// Row softmax over T=2048 (one block per (b,l) row).
// ---------------------------------------------------------------------------
__global__ __launch_bounds__(256) void softmax_kernel(float* __restrict__ out) {
    __shared__ float red[256];
    const int row = blockIdx.x;
    float* p = out + (size_t)row * TT;
    const int tid = threadIdx.x;
    float vals[8];
    float mx = -3.0e38f;
#pragma unroll
    for (int i = 0; i < 8; ++i) {
        vals[i] = p[tid + i * 256];
        mx = fmaxf(mx, vals[i]);
    }
    red[tid] = mx;
    __syncthreads();
    for (int s = 128; s > 0; s >>= 1) {
        if (tid < s) red[tid] = fmaxf(red[tid], red[tid + s]);
        __syncthreads();
    }
    mx = red[0];
    __syncthreads();
    float sum = 0.f;
#pragma unroll
    for (int i = 0; i < 8; ++i) {
        vals[i] = __expf(vals[i] - mx);
        sum += vals[i];
    }
    red[tid] = sum;
    __syncthreads();
    for (int s = 128; s > 0; s >>= 1) {
        if (tid < s) red[tid] += red[tid + s];
        __syncthreads();
    }
    float inv = 1.0f / red[0];
#pragma unroll
    for (int i = 0; i < 8; ++i) p[tid + i * 256] = vals[i] * inv;
}

extern "C" void kernel_launch(void* const* d_in, const int* in_sizes, int n_in,
                              void* d_out, int out_size, void* d_ws, size_t ws_size,
                              hipStream_t stream) {
    (void)in_sizes; (void)n_in; (void)out_size; (void)ws_size;
    const float* seq    = (const float*)d_in[0];
    const int*   slens  = (const int*)  d_in[1];
    /* d_in[2] = max_output_len (compile-time LL) */
    const float* Wp     = (const float*)d_in[3];
    const float* bp     = (const float*)d_in[4];
    const float* W_ih_e = (const float*)d_in[5];
    const float* W_hh_e = (const float*)d_in[6];
    const float* b_ih_e = (const float*)d_in[7];
    const float* b_hh_e = (const float*)d_in[8];
    const float* W_ih_d = (const float*)d_in[9];
    const float* W_hh_d = (const float*)d_in[10];
    const float* b_ih_d = (const float*)d_in[11];
    const float* b_hh_d = (const float*)d_in[12];
    const float* Wq     = (const float*)d_in[13];
    const float* bq     = (const float*)d_in[14];
    const float* Wk     = (const float*)d_in[15];
    const float* bk     = (const float*)d_in[16];
    float* out = (float*)d_out;

    char* ws = (char*)d_ws;
    size_t off = 0;
    auto alloc = [&](size_t bytes) -> char* {
        char* p = ws + off;
        off += (bytes + 255) & ~(size_t)255;
        return p;
    };
    float*    Weff = (float*)   alloc(2 * GG * sizeof(float));
    float*    beff = (float*)   alloc(GG * sizeof(float));
    float*    bd   = (float*)   alloc(GG * sizeof(float));
    uint32_t* WBe  = (uint32_t*)alloc((size_t)64 * 8 * 256 * 4);
    uint32_t* WBd  = (uint32_t*)alloc((size_t)64 * 8 * 256 * 4);
    uint32_t* WBk  = (uint32_t*)alloc((size_t)4 * 8 * 256 * 4);
    uint32_t* WBq  = (uint32_t*)alloc((size_t)4 * 8 * 256 * 4);
    uint16_t* kpP  = (uint16_t*)alloc((size_t)BB * 128 * 2 * 32 * 8 * 4);   // 8 MB
    float*    hT   = (float*)   alloc((size_t)BB * HH * sizeof(float));
    float*    cT   = (float*)   alloc((size_t)BB * HH * sizeof(float));
    uint16_t* Qb   = (uint16_t*)alloc((size_t)BB * LL * MM * 2);

    prep_misc<<<1, 1024, 0, stream>>>(Wp, bp, W_ih_e, b_ih_e, b_hh_e, b_ih_d, b_hh_d, Weff, beff, bd);
    pack_b<<<(64 * 8 * 256) / 256, 256, 0, stream>>>(W_hh_e, nullptr, WBe, 256, 1024, 0);
    pack_b<<<(64 * 8 * 256) / 256, 256, 0, stream>>>(W_ih_d, W_hh_d,  WBd, 256, 1024, 0);
    pack_b<<<(4 * 8 * 256) / 256, 256, 0, stream>>>(Wk, nullptr, WBk, 256, 64, 1);
    pack_b<<<(4 * 8 * 256) / 256, 256, 0, stream>>>(Wq, nullptr, WBq, 256, 64, 1);
    encoder_kernel<<<1, 1024, 0, stream>>>(seq, slens, WBe, Weff, beff, WBk, bk, kpP, hT, cT);
    decoder_kernel<<<1, 1024, 0, stream>>>(WBd, bd, WBq, bq, hT, cT, Qb);
    scores_kernel<<<dim3(8, 32), 1024, 0, stream>>>(Qb, kpP, slens, out);
    softmax_kernel<<<BB * LL, 256, 0, stream>>>(out);
}
